// fuse_block_Mamba_38835094291016
// MI455X (gfx1250) — matrix operand order repaired and checked by reading, not compiled, not run on hardware
//
#include <hip/hip_runtime.h>
#include <hip/hip_bf16.h>

// ---------------------------------------------------------------------------
// Vision-Mamba fused block for MI455X (gfx1250, wave32, WMMA bf16)
// ---------------------------------------------------------------------------

typedef __attribute__((ext_vector_type(16))) __bf16 v16bf;
typedef __attribute__((ext_vector_type(8)))  __bf16 v8bf;
typedef __attribute__((ext_vector_type(8)))  float  v8f;

#define BATCH   8
#define SEQ     392            // 2*196 tokens
#define DMODEL  768
#define DINNER  1536
#define DSTATE  16
#define MROWS   (BATCH*SEQ)    // 3136, multiple of 32

// -------------------------------- GEMM -------------------------------------
// C[M,N] = A[M,K] (bf16, row-major) x W[N,K]^T (bf16, K-contiguous)
//          (+ bias[N]) (+ resid[M,N]) (softplus epilogue optional)
// One wave computes a 32x32 macro-tile via 4x v_wmma_f32_16x16x32_bf16.

__device__ __forceinline__ v16bf join16(v8bf lo, v8bf hi) {
  return __builtin_shufflevector(lo, hi, 0,1,2,3,4,5,6,7,8,9,10,11,12,13,14,15);
}

__device__ __forceinline__ void store_tile(float* __restrict__ C,
                                           const float* __restrict__ bias,
                                           const float* __restrict__ resid,
                                           int N, int rbase, int cbase,
                                           int row_off, int col, v8f acc, int sp) {
#pragma unroll
  for (int i = 0; i < 8; ++i) {
    int rr = rbase + row_off + i;
    int cc = cbase + col;
    float v = acc[i];
    if (bias)  v += bias[cc];
    if (resid) v += resid[(size_t)rr * N + cc];
    if (sp)    v = (v > 20.f) ? v : log1pf(__expf(v));
    C[(size_t)rr * N + cc] = v;
  }
}

__global__ void gemm_wmma_bf16(const __bf16* __restrict__ A,
                               const __bf16* __restrict__ W,
                               const float* __restrict__ bias,
                               const float* __restrict__ resid,
                               float* __restrict__ C,
                               int M, int N, int K, int softplus_ep) {
  const int wave = blockIdx.x * (blockDim.x >> 5) + (threadIdx.x >> 5);
  const int lane = threadIdx.x & 31;
  const int tilesN = N >> 5;
  const int tilesM = M >> 5;
  if (wave >= tilesM * tilesN) return;
  const int tm = (wave / tilesN) << 5;
  const int tn = (wave % tilesN) << 5;

  const int half = lane >> 4;     // which 16-lane half of the wave
  const int r    = lane & 15;

  // A fragment rows (M = lane&15 for both halves; halves differ in K range)
  const __bf16* arow0 = A + (size_t)(tm + r) * K;
  const __bf16* arow1 = arow0 + (size_t)16 * K;
  // B fragment: lane -> output column, half -> K range same as A
  const __bf16* brow0 = W + (size_t)(tn + r) * K;
  const __bf16* brow1 = brow0 + (size_t)16 * K;

  v8f c00 = {}, c01 = {}, c10 = {}, c11 = {};

  for (int k0 = 0; k0 < K; k0 += 32) {
    const int ka = k0 + half * 8;
    __builtin_prefetch(arow0 + ka + 64, 0, 0);   // global_prefetch_b8
    __builtin_prefetch(brow0 + k0 + 64, 0, 0);

    v8bf a0lo = *(const v8bf*)(arow0 + ka);
    v8bf a0hi = *(const v8bf*)(arow0 + ka + 16);
    v8bf a1lo = *(const v8bf*)(arow1 + ka);
    v8bf a1hi = *(const v8bf*)(arow1 + ka + 16);
    v8bf b0lo = *(const v8bf*)(brow0 + ka);
    v8bf b0hi = *(const v8bf*)(brow0 + ka + 16);
    v8bf b1lo = *(const v8bf*)(brow1 + ka);
    v8bf b1hi = *(const v8bf*)(brow1 + ka + 16);

    v16bf a0 = join16(a0lo, a0hi);
    v16bf a1 = join16(a1lo, a1hi);
    v16bf b0 = join16(b0lo, b0hi);
    v16bf b1 = join16(b1lo, b1hi);

    c00 = __builtin_amdgcn_wmma_f32_16x16x32_bf16(false, a0, false, b0, (short)0, c00, false, false);
    c01 = __builtin_amdgcn_wmma_f32_16x16x32_bf16(false, a0, false, b1, (short)0, c01, false, false);
    c10 = __builtin_amdgcn_wmma_f32_16x16x32_bf16(false, a1, false, b0, (short)0, c10, false, false);
    c11 = __builtin_amdgcn_wmma_f32_16x16x32_bf16(false, a1, false, b1, (short)0, c11, false, false);
  }

  const int row_off = half * 8;   // C layout: lanes>=16 hold M+8 rows
  const int col     = lane & 15;
  store_tile(C, bias, resid, N, tm,      tn,      row_off, col, c00, softplus_ep);
  store_tile(C, bias, resid, N, tm,      tn + 16, row_off, col, c01, softplus_ep);
  store_tile(C, bias, resid, N, tm + 16, tn,      row_off, col, c10, softplus_ep);
  store_tile(C, bias, resid, N, tm + 16, tn + 16, row_off, col, c11, softplus_ep);
}

// ------------------------ f32 -> bf16 convert (+pad) -----------------------
__global__ void cvt_pad_bf16(const float* __restrict__ src, __bf16* __restrict__ dst,
                             int rs, int cs, int rd, int cd) {
  int idx = blockIdx.x * blockDim.x + threadIdx.x;
  int total = rd * cd;
  if (idx >= total) return;
  int r = idx / cd, c = idx % cd;
  float v = (r < rs && c < cs) ? src[(size_t)r * cs + c] : 0.f;
  dst[idx] = (__bf16)v;
}

// ------------------------------- im2col ------------------------------------
// A_pe[m, k]: m = b*392 + l (l<196 -> rgb patch, else thermal), k = (c,py,px)
__global__ void im2col_kernel(const float* __restrict__ rgb, const float* __restrict__ thm,
                              __bf16* __restrict__ ape) {
  int idx = blockIdx.x * blockDim.x + threadIdx.x;
  if (idx >= MROWS * DMODEL) return;
  int k = idx % DMODEL, m = idx / DMODEL;
  int b = m / SEQ, l = m % SEQ;
  int img = l / 196, p = l % 196;
  int ph = p / 14, pw = p % 14;
  int c = k / 256, rem = k % 256;
  int py = rem / 16, px = rem % 16;
  const float* src = img ? thm : rgb;
  float v = src[(((size_t)b * 3 + c) * 224 + ph * 16 + py) * 224 + pw * 16 + px];
  ape[idx] = (__bf16)v;
}

// ------------------------------- rmsnorm -----------------------------------
__global__ void rmsnorm_kernel(const float* __restrict__ x, const float* __restrict__ w,
                               __bf16* __restrict__ out) {
  __shared__ float red[256];
  int row = blockIdx.x;
  const float* xr = x + (size_t)row * DMODEL;
  float ss = 0.f;
  for (int c = threadIdx.x; c < DMODEL; c += blockDim.x) { float v = xr[c]; ss += v * v; }
  red[threadIdx.x] = ss;
  __syncthreads();
  for (int s = 128; s > 0; s >>= 1) {
    if (threadIdx.x < s) red[threadIdx.x] += red[threadIdx.x + s];
    __syncthreads();
  }
  float scale = rsqrtf(red[0] / (float)DMODEL + 1e-5f);
  __bf16* o = out + (size_t)row * DMODEL;
  for (int c = threadIdx.x; c < DMODEL; c += blockDim.x)
    o[c] = (__bf16)(xr[c] * scale * w[c]);
}

// --------------------- depthwise causal conv + SiLU ------------------------
__global__ void conv_silu_kernel(const float* __restrict__ xz, const float* __restrict__ cw,
                                 const float* __restrict__ cb, float* __restrict__ xact,
                                 __bf16* __restrict__ xbf) {
  int idx = blockIdx.x * blockDim.x + threadIdx.x;
  if (idx >= MROWS * DINNER) return;
  int e = idx % DINNER, m = idx / DINNER;
  int l = m % SEQ;
  float acc = cb[e];
#pragma unroll
  for (int j = 0; j < 4; ++j) {
    int ll = l - 3 + j;
    if (ll >= 0) acc += cw[e * 4 + j] * xz[(size_t)(m - (l - ll)) * (2 * DINNER) + e];
  }
  float s = acc * (1.f / (1.f + __expf(-acc)));
  xact[idx] = s;
  xbf[idx]  = (__bf16)s;
}

// ----------------- build padded bf16 delta_low [M,64] from dbc -------------
__global__ void dlow_kernel(const float* __restrict__ dbc, __bf16* __restrict__ dlow) {
  int idx = blockIdx.x * blockDim.x + threadIdx.x;
  if (idx >= MROWS * 64) return;
  int c = idx & 63, m = idx >> 6;
  float v = (c < 48) ? dbc[(size_t)m * 96 + c] : 0.f;
  dlow[idx] = (__bf16)v;
}

// --------------------------- selective scan --------------------------------
// One thread per (b, e): 16-state recurrence over L=392; coalesced over e.
__global__ void scan_kernel(const float* __restrict__ delta, const float* __restrict__ xact,
                            const float* __restrict__ dbc, const float* __restrict__ xz,
                            const float* __restrict__ a_log, const float* __restrict__ dp,
                            __bf16* __restrict__ ybf) {
  int tid = blockIdx.x * blockDim.x + threadIdx.x;
  if (tid >= BATCH * DINNER) return;
  int e = tid % DINNER, b = tid / DINNER;
  float An[DSTATE], h[DSTATE];
#pragma unroll
  for (int n = 0; n < DSTATE; ++n) { An[n] = -__expf(a_log[e * DSTATE + n]); h[n] = 0.f; }
  float De = dp[e];
  for (int l = 0; l < SEQ; ++l) {
    int m = b * SEQ + l;
    size_t me = (size_t)m * DINNER + e;
    float dt  = delta[me];
    float xv  = xact[me];
    float dtx = dt * xv;
    const float* bc = dbc + (size_t)m * 96;
    float yv = 0.f;
#pragma unroll
    for (int n = 0; n < DSTATE; ++n) {
      float da = __expf(dt * An[n]);
      h[n] = da * h[n] + dtx * bc[48 + n];
      yv += h[n] * bc[64 + n];
    }
    yv += De * xv;
    float zv = xz[(size_t)m * (2 * DINNER) + DINNER + e];
    yv *= zv * (1.f / (1.f + __expf(-zv)));
    ybf[me] = (__bf16)yv;
  }
}

// ---------------------------- final permute --------------------------------
// out[(l*768 + d)*8 + b] = x[(b*392 + l)*768 + d]
__global__ void permute_out_kernel(const float* __restrict__ x, float* __restrict__ out) {
  int idx = blockIdx.x * blockDim.x + threadIdx.x;
  if (idx >= MROWS * DMODEL) return;
  int b = idx & 7;
  int d = (idx >> 3) % DMODEL;
  int l = idx / (8 * DMODEL);
  out[idx] = x[((size_t)b * SEQ + l) * DMODEL + d];
}

// ---------------------------------------------------------------------------
extern "C" void kernel_launch(void* const* d_in, const int* in_sizes, int n_in,
                              void* d_out, int out_size, void* d_ws, size_t ws_size,
                              hipStream_t stream) {
  const float* rgb   = (const float*)d_in[0];
  const float* thm   = (const float*)d_in[1];
  const float* pe_w  = (const float*)d_in[2];
  const float* pe_b  = (const float*)d_in[3];
  const float* inw   = (const float*)d_in[4];
  const float* cw    = (const float*)d_in[5];
  const float* cb    = (const float*)d_in[6];
  const float* xpw   = (const float*)d_in[7];
  const float* dtw   = (const float*)d_in[8];
  const float* dtb   = (const float*)d_in[9];
  const float* alog  = (const float*)d_in[10];
  const float* dpar  = (const float*)d_in[11];
  const float* outw  = (const float*)d_in[12];
  const float* nw    = (const float*)d_in[13];

  char* ws = (char*)d_ws;
  size_t off = 0;
  auto alloc = [&](size_t bytes) -> char* {
    char* p = ws + off;
    off = (off + bytes + 255) & ~(size_t)255;
    return p;
  };

  float*  x0    = (float*) alloc((size_t)MROWS * DMODEL * 4);
  float*  x1    = (float*) alloc((size_t)MROWS * DMODEL * 4);
  __bf16* xn    = (__bf16*)alloc((size_t)MROWS * DMODEL * 2);
  __bf16* ape   = (__bf16*)alloc((size_t)MROWS * DMODEL * 2);
  float*  xz    = (float*) alloc((size_t)MROWS * 2 * DINNER * 4);
  float*  xact  = (float*) alloc((size_t)MROWS * DINNER * 4);
  __bf16* xbf   = (__bf16*)alloc((size_t)MROWS * DINNER * 2);
  float*  dbc   = (float*) alloc((size_t)MROWS * 96 * 4);
  __bf16* dlow  = (__bf16*)alloc((size_t)MROWS * 64 * 2);
  float*  delta = (float*) alloc((size_t)MROWS * DINNER * 4);
  __bf16* ybf   = (__bf16*)alloc((size_t)MROWS * DINNER * 2);
  __bf16* wpe   = (__bf16*)alloc((size_t)DMODEL * DMODEL * 2);
  __bf16* win   = (__bf16*)alloc((size_t)2 * DINNER * DMODEL * 2);
  __bf16* wx    = (__bf16*)alloc((size_t)96 * DINNER * 2);
  __bf16* wdt   = (__bf16*)alloc((size_t)DINNER * 64 * 2);
  __bf16* wout  = (__bf16*)alloc((size_t)DMODEL * DINNER * 2);

  auto blocks = [](long long total, int bs) { return (int)((total + bs - 1) / bs); };
  auto gemm_grid = [](int M, int N) { int waves = (M >> 5) * (N >> 5); return (waves + 7) / 8; };

  // ---- patch embed ----
  cvt_pad_bf16<<<blocks((long long)DMODEL * DMODEL, 256), 256, 0, stream>>>(
      pe_w, wpe, DMODEL, DMODEL, DMODEL, DMODEL);
  im2col_kernel<<<blocks((long long)MROWS * DMODEL, 256), 256, 0, stream>>>(rgb, thm, ape);
  gemm_wmma_bf16<<<gemm_grid(MROWS, DMODEL), 256, 0, stream>>>(
      ape, wpe, pe_b, nullptr, x0, MROWS, DMODEL, DMODEL, 0);

  float* xcur = x0;
  float* xnxt = x1;

  for (int i = 0; i < 8; ++i) {
    const float* inw_i  = inw  + (size_t)i * 2 * DINNER * DMODEL;
    const float* cw_i   = cw   + (size_t)i * DINNER * 4;
    const float* cb_i   = cb   + (size_t)i * DINNER;
    const float* xpw_i  = xpw  + (size_t)i * 80 * DINNER;
    const float* dtw_i  = dtw  + (size_t)i * DINNER * 48;
    const float* dtb_i  = dtb  + (size_t)i * DINNER;
    const float* alog_i = alog + (size_t)i * DINNER * DSTATE;
    const float* d_i    = dpar + (size_t)i * DINNER;
    const float* outw_i = outw + (size_t)i * DMODEL * DINNER;
    const float* nw_i   = nw   + (size_t)i * DMODEL;

    // rmsnorm -> bf16
    rmsnorm_kernel<<<MROWS, 256, 0, stream>>>(xcur, nw_i, xn);

    // in_proj: [M,768] x [3072,768]^T -> xz [M,3072]
    cvt_pad_bf16<<<blocks((long long)2 * DINNER * DMODEL, 256), 256, 0, stream>>>(
        inw_i, win, 2 * DINNER, DMODEL, 2 * DINNER, DMODEL);
    gemm_wmma_bf16<<<gemm_grid(MROWS, 2 * DINNER), 256, 0, stream>>>(
        xn, win, nullptr, nullptr, xz, MROWS, 2 * DINNER, DMODEL, 0);

    // depthwise causal conv + SiLU
    conv_silu_kernel<<<blocks((long long)MROWS * DINNER, 256), 256, 0, stream>>>(
        xz, cw_i, cb_i, xact, xbf);

    // x_proj: [M,1536] x [80,1536]^T (padded N->96) -> dbc [M,96]
    cvt_pad_bf16<<<blocks((long long)96 * DINNER, 256), 256, 0, stream>>>(
        xpw_i, wx, 80, DINNER, 96, DINNER);
    gemm_wmma_bf16<<<gemm_grid(MROWS, 96), 256, 0, stream>>>(
        xbf, wx, nullptr, nullptr, dbc, MROWS, 96, DINNER, 0);

    // dt_proj: [M,48->64] x [1536,48->64]^T + bias, softplus -> delta
    dlow_kernel<<<blocks((long long)MROWS * 64, 256), 256, 0, stream>>>(dbc, dlow);
    cvt_pad_bf16<<<blocks((long long)DINNER * 64, 256), 256, 0, stream>>>(
        dtw_i, wdt, DINNER, 48, DINNER, 64);
    gemm_wmma_bf16<<<gemm_grid(MROWS, DINNER), 256, 0, stream>>>(
        dlow, wdt, dtb_i, nullptr, delta, MROWS, DINNER, 64, 1);

    // selective scan + D-skip + SiLU gate -> y (bf16)
    scan_kernel<<<blocks((long long)BATCH * DINNER, 256), 256, 0, stream>>>(
        delta, xact, dbc, xz, alog_i, d_i, ybf);

    // out_proj + residual: [M,1536] x [768,1536]^T + xcur -> xnxt
    cvt_pad_bf16<<<blocks((long long)DMODEL * DINNER, 256), 256, 0, stream>>>(
        outw_i, wout, DMODEL, DINNER, DMODEL, DINNER);
    gemm_wmma_bf16<<<gemm_grid(MROWS, DMODEL), 256, 0, stream>>>(
        ybf, wout, nullptr, xcur, xnxt, MROWS, DMODEL, DINNER, 0);

    float* t = xcur; xcur = xnxt; xnxt = t;
  }

  // final permute to output ([1,392,768,B] row-major flat)
  permute_out_kernel<<<blocks((long long)MROWS * DMODEL, 256), 256, 0, stream>>>(
      xcur, (float*)d_out);
}
